// DualLSTMDecoder_75496935129225
// MI455X (gfx1250) — matrix operand order repaired and checked by reading, not compiled, not run on hardware
//
#include <hip/hip_runtime.h>
#include <hip/hip_bf16.h>

typedef __attribute__((ext_vector_type(16))) _Float16 v16h;
typedef __attribute__((ext_vector_type(8)))  _Float16 v8h;
typedef __attribute__((ext_vector_type(4)))  _Float16 v4h;
typedef __attribute__((ext_vector_type(8)))  float    v8f;

#define B_  256
#define L_  512
#define E_  256
#define H_  128
#define G_  512   // 4*H

// workspace layout (bytes)
#define WS_WIH   0u          // 2 lstm * 512*256 halves  = 524288 B
#define WS_WHH   524288u     // 2 lstm * 512*128 halves  = 262144 B
#define WS_BSUM  786432u     // 2 lstm * 512 floats      = 4096 B
#define WS_PART  790528u     // 2 lstm * 256*512 floats  = 1048576 B
#define WS_GX    1839104u    // 2 lstm * 256*512*512 halves = 268435456 B
#define WS_NEED  (WS_GX + (size_t)2 * B_ * L_ * G_ * 2)

// ---------------------------------------------------------------------------
// Kernel 0: convert weights f32 -> f16 into workspace, fold biases.
// ---------------------------------------------------------------------------
__global__ void prep_kernel(const float* __restrict__ Wih_h, const float* __restrict__ Whh_h,
                            const float* __restrict__ bih_h, const float* __restrict__ bhh_h,
                            const float* __restrict__ Wih_f, const float* __restrict__ Whh_f,
                            const float* __restrict__ bih_f, const float* __restrict__ bhh_f,
                            _Float16* __restrict__ wih16, _Float16* __restrict__ whh16,
                            float* __restrict__ bsum)
{
    const int stride = gridDim.x * blockDim.x;
    for (int i = blockIdx.x * blockDim.x + threadIdx.x; i < G_ * E_; i += stride) {
        wih16[i]            = (_Float16)Wih_h[i];
        wih16[G_ * E_ + i]  = (_Float16)Wih_f[i];
        if (i < G_ * H_) {
            whh16[i]           = (_Float16)Whh_h[i];
            whh16[G_ * H_ + i] = (_Float16)Whh_f[i];
        }
        if (i < G_) {
            bsum[i]      = bih_h[i] + bhh_h[i];
            bsum[G_ + i] = bih_f[i] + bhh_f[i];
        }
    }
}

// ---------------------------------------------------------------------------
// Kernel 1a: bulk GEMM  gx[row, n] = bias[n] + x[row,:] @ Wih[n,:]   (f16 out)
// row = b*L + t  (x is row-major [B*L][E], gx is row-major [B*L][G]).
// grid = 1024 blocks x 256 thr (8 waves).  Decode: lstm | nquad(4) | mchunk(128).
// Wave w owns N-tile nq*8+w; its 8 W_ih B-fragments stay in VGPRs across the
// block's 64 M-tiles, so weights are read from L2 once per block (~64 MB total).
// ---------------------------------------------------------------------------
__global__ __launch_bounds__(256)
void gx_kernel(const float* __restrict__ hist, const float* __restrict__ fut,
               const _Float16* __restrict__ wih16, const float* __restrict__ bsum,
               _Float16* __restrict__ gx)
{
    __shared__ __align__(16) unsigned char smem[8192 + 16384];
    _Float16* sh_x  = (_Float16*)(smem);           // [16][256] f16 A-stage
    _Float16* sh_gx = (_Float16*)(smem + 8192);    // [16][512] f16 C-stage

    const int  tid  = threadIdx.x;
    const int  wave = tid >> 5;
    const int  lane = tid & 31;
    const int  ln   = lane & 15;
    const bool hi   = lane >= 16;

    const int lstm = blockIdx.x >> 9;
    const int nq   = (blockIdx.x >> 7) & 3;
    const int mc   = blockIdx.x & 127;

    const float* __restrict__ xsrc = lstm ? fut : hist;
    const _Float16* __restrict__ wihp = wih16 + (size_t)lstm * (G_ * E_);
    const float* bs = bsum + lstm * G_;
    _Float16* __restrict__ gxp = gx + (size_t)lstm * B_ * L_ * G_;

    const int  n    = (nq * 8 + wave) * 16 + ln;   // this lane's gate column
    const int  ksel = hi ? 16 : 0;
    const int  klo  = hi ? 8 : 0;
    const float bsv = bs[n];

    v16h wib[8];                                   // W_ih fragments (64 VGPRs)
    #pragma unroll
    for (int kt = 0; kt < 8; ++kt)
        wib[kt] = *(const v16h*)(wihp + (size_t)n * E_ + kt * 32 + ksel);

    for (int i = 0; i < 64; ++i) {                 // 64 M-tiles per block
        const size_t m = (size_t)mc * 64 + i;      // global M-tile, rows m*16..+15

        // stage A: [16 x 256] f32 -> f16 (rows are contiguous in x)
        #pragma unroll
        for (int q = 0; q < 4; ++q) {
            const int id = tid + 256 * q;
            const int r  = id >> 6;
            const int c4 = id & 63;
            const float4 xv = *(const float4*)(xsrc + (m * 16 + r) * E_ + c4 * 4);
            v4h hx = { (_Float16)xv.x, (_Float16)xv.y, (_Float16)xv.z, (_Float16)xv.w };
            *(v4h*)(sh_x + r * E_ + c4 * 4) = hx;
            if (i + 1 < 64)
                __builtin_prefetch(xsrc + (m * 16 + 16 + r) * E_ + c4 * 4, 0, 1);
        }
        __syncthreads();

        v8f acc;
        #pragma unroll
        for (int r = 0; r < 8; ++r) acc[r] = bsv;
        #pragma unroll
        for (int kt = 0; kt < 8; ++kt) {
            v8h alo = *(const v8h*)(sh_x + ln * E_ + kt * 32 + ksel);
            v8h ahi = *(const v8h*)(sh_x + ln * E_ + kt * 32 + ksel + 8);
            v16h a;
            #pragma unroll
            for (int e = 0; e < 8; ++e) { a[e] = alo[e]; a[e + 8] = ahi[e]; }
            acc = __builtin_amdgcn_wmma_f32_16x16x32_f16(
                false, a, false, wib[kt], (short)0, acc, false, false);
        }
        // dump C tile to LDS (C layout: VGPR r <-> row r+8*hi, lane <-> N)
        #pragma unroll
        for (int r = 0; r < 8; ++r)
            sh_gx[(r + (hi ? 8 : 0)) * G_ + n] = (_Float16)acc[r];
        __syncthreads();

        // coalesced b128 store of the contiguous 16 KB tile
        #pragma unroll
        for (int q = 0; q < 4; ++q) {
            const int id = tid + 256 * q;
            *(v8h*)(gxp + m * (16 * G_) + id * 8) = *(const v8h*)(sh_gx + id * 8);
        }
        __syncthreads();
    }
}

// ---------------------------------------------------------------------------
// Kernel 1b: recurrence only:  gates = gx(t) + h @ Whh^T, then cell update.
// grid = 32 blocks (lstm | batch-block), 1024 thr = 32 waves, 1 N-tile/wave.
// Register demand ~70 VGPRs << 128 cap: no spills; W_hh VGPR-resident.
// ---------------------------------------------------------------------------
__global__ __launch_bounds__(1024)
void rec_kernel(const _Float16* __restrict__ gx, const _Float16* __restrict__ whh16,
                const float* __restrict__ wffn, float* __restrict__ part)
{
    __shared__ __align__(16) unsigned char smem[53760];
    float*    sh_red = (float*)(smem);             // [2048] f32  (8192 B)
    _Float16* sh_h   = (_Float16*)(smem + 8192);   // [16][128] f16
    float*    sh_c   = (float*)(smem + 12288);     // [16][128] f32
    float*    sh_g   = (float*)(smem + 20480);     // [16][512] f32
    float*    sh_w   = (float*)(smem + 53248);     // [128] f32

    const int  tid  = threadIdx.x;
    const int  wave = tid >> 5;
    const int  lane = tid & 31;
    const int  ln   = lane & 15;
    const bool hi   = lane >= 16;

    const int lstm = blockIdx.x >> 4;
    const int b0   = (blockIdx.x & 15) * 16;

    const _Float16* __restrict__ gxp = gx + (size_t)lstm * B_ * L_ * G_;
    const _Float16* __restrict__ whh = whh16 + (size_t)lstm * (G_ * H_);
    const float* wf = wffn + lstm * H_;
    float*       pb = part + (size_t)lstm * (B_ * L_);

    for (int i = tid; i < 16 * H_; i += 1024) { sh_h[i] = (_Float16)0.f; sh_c[i] = 0.f; }
    if (tid < H_) sh_w[tid] = wf[tid];

    const int n    = wave * 16 + ln;
    const int ksel = hi ? 16 : 0;
    const int klo  = hi ? 8 : 0;
    const int m8   = hi ? 8 : 0;

    v16h whf[4];                                   // W_hh resident (32 VGPRs)
    #pragma unroll
    for (int kt = 0; kt < 4; ++kt)
        whf[kt] = *(const v16h*)(whh + (size_t)n * H_ + kt * 32 + ksel);

    __syncthreads();

    for (int t = 0; t < L_; ++t) {
        // ---- accumulator starts from precomputed gates_x (bias folded in)
        v8f acc;
        #pragma unroll
        for (int r = 0; r < 8; ++r)
            acc[r] = (float)gxp[((size_t)(b0 + r + m8) * L_ + t) * G_ + n];
        if (t + 1 < L_)
            __builtin_prefetch(gxp + ((size_t)(b0 + m8) * L_ + (t + 1)) * G_ + n, 0, 1);

        // ---- + h @ Whh^T  (4 WMMAs, everything on-chip)
        #pragma unroll
        for (int kt = 0; kt < 4; ++kt) {
            v8h alo = *(const v8h*)(sh_h + ln * H_ + kt * 32 + ksel);
            v8h ahi = *(const v8h*)(sh_h + ln * H_ + kt * 32 + ksel + 8);
            v16h a;
            #pragma unroll
            for (int e = 0; e < 8; ++e) { a[e] = alo[e]; a[e + 8] = ahi[e]; }
            acc = __builtin_amdgcn_wmma_f32_16x16x32_f16(
                false, a, false, whf[kt], (short)0, acc, false, false);
        }
        #pragma unroll
        for (int r = 0; r < 8; ++r)
            sh_g[(r + m8) * G_ + n] = acc[r];
        __syncthreads();

        // ---- elementwise LSTM cell + FFN partial products
        #pragma unroll
        for (int q = 0; q < 2; ++q) {
            const int p = tid + 1024 * q;          // p = r*128 + j
            const int j = p & (H_ - 1);
            const int r = p >> 7;
            const float gi = sh_g[r * G_ + j];
            const float gf = sh_g[r * G_ + H_ + j];
            const float gg = sh_g[r * G_ + 2 * H_ + j];
            const float go = sh_g[r * G_ + 3 * H_ + j];
            const float ii = 1.f / (1.f + __expf(-gi));
            const float ff = 1.f / (1.f + __expf(-gf));
            const float g  = 1.f - 2.f / (__expf(2.f * gg) + 1.f);   // tanh, inf-safe
            const float oo = 1.f / (1.f + __expf(-go));
            const float c  = ff * sh_c[r * H_ + j] + ii * g;
            sh_c[r * H_ + j] = c;
            const float h  = oo * (1.f - 2.f / (__expf(2.f * c) + 1.f));
            sh_h[r * H_ + j] = (_Float16)h;
            sh_red[p] = h * sh_w[j];
        }
        __syncthreads();

        if (tid < 16) {
            float s = 0.f;
            #pragma unroll 8
            for (int j = 0; j < H_; ++j) s += sh_red[tid * H_ + j];
            pb[(size_t)(b0 + tid) * L_ + t] = s;
        }
        __syncthreads();
    }
}

// ---------------------------------------------------------------------------
// Kernel 1c (fallback when ws is too small for gx): fused persistent LSTM.
// 4 resident + 4 streamed W_ih fragments; W_hh resident.
// ---------------------------------------------------------------------------
__global__ __launch_bounds__(1024)
void lstm_fused_kernel(const float* __restrict__ hist, const float* __restrict__ fut,
                       const _Float16* __restrict__ wih16, const _Float16* __restrict__ whh16,
                       const float* __restrict__ bsum, const float* __restrict__ wffn,
                       float* __restrict__ part)
{
    __shared__ __align__(16) unsigned char smem[53760];
    _Float16* sh_x   = (_Float16*)(smem);
    _Float16* sh_h   = (_Float16*)(smem + 8192);
    float*    sh_c   = (float*)(smem + 12288);
    float*    sh_g   = (float*)(smem + 20480);
    float*    sh_w   = (float*)(smem + 53248);
    float*    sh_red = (float*)(smem);

    const int  tid  = threadIdx.x;
    const int  wave = tid >> 5;
    const int  lane = tid & 31;
    const int  ln   = lane & 15;
    const bool hi   = lane >= 16;

    const int lstm = blockIdx.x >> 4;
    const int b0   = (blockIdx.x & 15) * 16;

    const float*    __restrict__ xsrc = lstm ? fut : hist;
    const _Float16* __restrict__ wih  = wih16 + (size_t)lstm * (G_ * E_);
    const _Float16* __restrict__ whh  = whh16 + (size_t)lstm * (G_ * H_);
    const float* bs = bsum + lstm * G_;
    const float* wf = wffn + lstm * H_;
    float*       pb = part + (size_t)lstm * (B_ * L_);

    for (int i = tid; i < 16 * H_; i += 1024) { sh_h[i] = (_Float16)0.f; sh_c[i] = 0.f; }
    if (tid < H_) sh_w[tid] = wf[tid];

    const int n    = wave * 16 + ln;
    const float bsv = bs[n];
    const int ksel = hi ? 16 : 0;
    const int klo  = hi ? 8 : 0;
    const int m8   = hi ? 8 : 0;
    const _Float16* __restrict__ wihn = wih + (size_t)n * E_ + ksel;

    v16h wif[4];
    #pragma unroll
    for (int kt = 0; kt < 4; ++kt)
        wif[kt] = *(const v16h*)(wihn + kt * 32);
    v16h whf[4];
    #pragma unroll
    for (int kt = 0; kt < 4; ++kt)
        whf[kt] = *(const v16h*)(whh + (size_t)n * H_ + kt * 32 + ksel);
    __syncthreads();

    for (int t = 0; t < L_; ++t) {
        {
            const int r  = tid >> 6;
            const int c4 = tid & 63;
            const float4 xv = *(const float4*)(xsrc + ((size_t)(b0 + r) * L_ + t) * E_ + c4 * 4);
            v4h hx = { (_Float16)xv.x, (_Float16)xv.y, (_Float16)xv.z, (_Float16)xv.w };
            *(v4h*)(sh_x + r * E_ + c4 * 4) = hx;
            if (t + 1 < L_)
                __builtin_prefetch(xsrc + ((size_t)(b0 + r) * L_ + (t + 1)) * E_ + c4 * 4, 0, 1);
        }
        __syncthreads();

        // stream W_ih kt=4..7 from L2 each step; laundered offset defeats LICM
        int off4 = 4 * 32;
        asm volatile("" : "+v"(off4));
        v16h ws0 = *(const v16h*)(wihn + off4);
        v16h ws1 = *(const v16h*)(wihn + off4 + 32);
        v16h ws2, ws3;

        v8f acc;
        #pragma unroll
        for (int r = 0; r < 8; ++r) acc[r] = bsv;

        #pragma unroll
        for (int kt = 0; kt < 8; ++kt) {
            v8h alo = *(const v8h*)(sh_x + ln * E_ + kt * 32 + ksel);
            v8h ahi = *(const v8h*)(sh_x + ln * E_ + kt * 32 + ksel + 8);
            v16h a;
            #pragma unroll
            for (int e = 0; e < 8; ++e) { a[e] = alo[e]; a[e + 8] = ahi[e]; }
            if (kt == 4) {
                ws2 = *(const v16h*)(wihn + off4 + 64);
                ws3 = *(const v16h*)(wihn + off4 + 96);
            }
            const v16h bf = (kt < 4) ? wif[kt]
                          : (kt == 4) ? ws0 : (kt == 5) ? ws1 : (kt == 6) ? ws2 : ws3;
            acc = __builtin_amdgcn_wmma_f32_16x16x32_f16(
                false, a, false, bf, (short)0, acc, false, false);
        }
        #pragma unroll
        for (int kt = 0; kt < 4; ++kt) {
            v8h alo = *(const v8h*)(sh_h + ln * H_ + kt * 32 + ksel);
            v8h ahi = *(const v8h*)(sh_h + ln * H_ + kt * 32 + ksel + 8);
            v16h a;
            #pragma unroll
            for (int e = 0; e < 8; ++e) { a[e] = alo[e]; a[e + 8] = ahi[e]; }
            acc = __builtin_amdgcn_wmma_f32_16x16x32_f16(
                false, a, false, whf[kt], (short)0, acc, false, false);
        }
        #pragma unroll
        for (int r = 0; r < 8; ++r)
            sh_g[(r + m8) * G_ + n] = acc[r];
        __syncthreads();

        #pragma unroll
        for (int q = 0; q < 2; ++q) {
            const int p = tid + 1024 * q;
            const int j = p & (H_ - 1);
            const int r = p >> 7;
            const float gi = sh_g[r * G_ + j];
            const float gf = sh_g[r * G_ + H_ + j];
            const float gg = sh_g[r * G_ + 2 * H_ + j];
            const float go = sh_g[r * G_ + 3 * H_ + j];
            const float ii = 1.f / (1.f + __expf(-gi));
            const float ff = 1.f / (1.f + __expf(-gf));
            const float g  = 1.f - 2.f / (__expf(2.f * gg) + 1.f);
            const float oo = 1.f / (1.f + __expf(-go));
            const float c  = ff * sh_c[r * H_ + j] + ii * g;
            sh_c[r * H_ + j] = c;
            const float h  = oo * (1.f - 2.f / (__expf(2.f * c) + 1.f));
            sh_h[r * H_ + j] = (_Float16)h;
            sh_red[p] = h * sh_w[j];
        }
        __syncthreads();

        if (tid < 16) {
            float s = 0.f;
            #pragma unroll 8
            for (int j = 0; j < H_; ++j) s += sh_red[tid * H_ + j];
            pb[(size_t)(b0 + tid) * L_ + t] = s;
        }
        __syncthreads();
    }
}

// ---------------------------------------------------------------------------
// Kernel 2: out[b,t] = part_hist[b,t] + part_fut[b,t] + b_ffn
// ---------------------------------------------------------------------------
__global__ void fin_kernel(const float* __restrict__ part, const float* __restrict__ bffn,
                           float* __restrict__ out)
{
    const int i = blockIdx.x * blockDim.x + threadIdx.x;
    if (i < B_ * L_)
        out[i] = part[i] + part[B_ * L_ + i] + bffn[0];
}

extern "C" void kernel_launch(void* const* d_in, const int* in_sizes, int n_in,
                              void* d_out, int out_size, void* d_ws, size_t ws_size,
                              hipStream_t stream)
{
    const float* hist  = (const float*)d_in[0];
    const float* fut   = (const float*)d_in[1];
    const float* Wih_h = (const float*)d_in[2];
    const float* Whh_h = (const float*)d_in[3];
    const float* bih_h = (const float*)d_in[4];
    const float* bhh_h = (const float*)d_in[5];
    const float* Wih_f = (const float*)d_in[6];
    const float* Whh_f = (const float*)d_in[7];
    const float* bih_f = (const float*)d_in[8];
    const float* bhh_f = (const float*)d_in[9];
    const float* Wffn  = (const float*)d_in[10];
    const float* bffn  = (const float*)d_in[11];
    float* out = (float*)d_out;

    unsigned char* ws = (unsigned char*)d_ws;
    _Float16* wih16 = (_Float16*)(ws + WS_WIH);
    _Float16* whh16 = (_Float16*)(ws + WS_WHH);
    float*    bsum  = (float*)(ws + WS_BSUM);
    float*    part  = (float*)(ws + WS_PART);
    _Float16* gx    = (_Float16*)(ws + WS_GX);

    hipLaunchKernelGGL(prep_kernel, dim3(512), dim3(256), 0, stream,
                       Wih_h, Whh_h, bih_h, bhh_h, Wih_f, Whh_f, bih_f, bhh_f,
                       wih16, whh16, bsum);

    if (ws_size >= WS_NEED) {
        // split path: bulk WMMA GEMM for gates_x, then lean recurrence
        hipLaunchKernelGGL(gx_kernel, dim3(1024), dim3(256), 0, stream,
                           hist, fut, wih16, bsum, gx);
        hipLaunchKernelGGL(rec_kernel, dim3(32), dim3(1024), 0, stream,
                           gx, whh16, Wffn, part);
    } else {
        // fallback: fused persistent recurrence
        hipLaunchKernelGGL(lstm_fused_kernel, dim3(32), dim3(1024), 0, stream,
                           hist, fut, wih16, whh16, bsum, Wffn, part);
    }
    hipLaunchKernelGGL(fin_kernel, dim3(512), dim3(256), 0, stream,
                       part, bffn, out);
}
